// XCAttention_16947940950097
// MI455X (gfx1250) — matrix operand order repaired and checked by reading, not compiled, not run on hardware
//
#include <hip/hip_runtime.h>
#include <hip/hip_bf16.h>
#include <math.h>

typedef __bf16 bf16;
typedef __attribute__((ext_vector_type(16))) __bf16        v16bf;
typedef __attribute__((ext_vector_type(8)))  float         v8f;
typedef __attribute__((ext_vector_type(4)))  unsigned int  v4u;
typedef __attribute__((ext_vector_type(4)))  float         v4f;
typedef __attribute__((ext_vector_type(4)))  int           v4i;

#define BB 16
#define SS 3136
#define EE 768
#define HH 16
#define DD 48
#define XEPS 1e-12f

// CDNA5 async global->LDS path (ASYNCcnt-tracked), guarded so the file
// still compiles if the toolchain does not expose these builtins.
#if defined(__has_builtin)
#  if __has_builtin(__builtin_amdgcn_global_load_async_to_lds_b128) && \
      __has_builtin(__builtin_amdgcn_s_wait_asynccnt)
#    define USE_ASYNC_LDS 1
#  endif
#endif
#ifndef USE_ASYNC_LDS
#  define USE_ASYNC_LDS 0
#endif

#if USE_ASYNC_LDS
typedef __attribute__((address_space(1))) v4i* as1_v4i;
typedef __attribute__((address_space(3))) v4i* as3_v4i;
#endif

union FragU { v16bf v; v4u q[2]; };
union Pack8 { v4u q; bf16 h[8]; };

__device__ __forceinline__ v8f wmma_bf16(v16bf a, v16bf b, v8f c) {
  // D = A(16x32 bf16) * B(32x16 bf16) + C(16x16 f32)
  return __builtin_amdgcn_wmma_f32_16x16x32_bf16(false, a, false, b, (short)0, c,
                                                 false, false);
}

// A fragment: row-major [M][K], row stride ld (elements).
// lane l: row = l&15 ; lo lanes hold K 0-7 / 16-23, hi lanes K 8-15 / 24-31.
__device__ __forceinline__ v16bf frag_a(const bf16* p, int ld) {
  int lane = threadIdx.x & 31;
  int row  = lane & 15;
  int kh   = (lane >> 4) << 3;   // 0 or 8
  FragU f;
  f.q[0] = *(const v4u*)(p + row * ld + kh);
  f.q[1] = *(const v4u*)(p + row * ld + kh + 16);
  return f.v;
}

// B fragment from [N][K] layout (K contiguous per row), row stride ld.
// lane l: col = l&15 ; lo lanes hold K 0-7 / 16-23, hi lanes K 8-15 / 24-31.
__device__ __forceinline__ v16bf frag_b(const bf16* p, int ld) {
  int lane = threadIdx.x & 31;
  int col  = lane & 15;
  int kb   = (lane >> 4) << 3;   // 0 or 8
  FragU f;
  f.q[0] = *(const v4u*)(p + col * ld + kb);
  f.q[1] = *(const v4u*)(p + col * ld + kb + 16);
  return f.v;
}

// ---------------------------------------------------------------------------
// Kernel 1: qkv = x @ W_qkv -> bf16 transposed layout qkvt[which][b][h][d][s].
// BM=128 BN=64 BK=32, 8 waves, 2x2 C tiles per wave (4 WMMA / 4 frag loads).
// ---------------------------------------------------------------------------
__global__ __launch_bounds__(256) void k_qkv(const float* __restrict__ x,
                                             const float* __restrict__ w,
                                             bf16* __restrict__ qkvt) {
  __shared__ bf16 As[128 * 40];   // [m][k] pad->40
  __shared__ bf16 Bt[64 * 40];    // [n][k] pad->40
  __shared__ bf16 Cs[64 * 136];   // [n][m] transpose bounce (pad->136)

  const int tid  = threadIdx.x;
  const int wv   = tid >> 5;
  const int lane = tid & 31;
  const int m0 = blockIdx.x * 128;       // row in [B*S]
  const int n0 = blockIdx.y * 64;        // col in [3E]
  const int mi0 = (wv & 3) * 2;          // M tile pair: 0,2,4,6
  const int nj0 = (wv >> 2) * 2;         // N tile pair: 0 or 2

  v8f a00 = {}, a01 = {}, a10 = {}, a11 = {};

  const int arow = tid >> 1, akc = (tid & 1) * 16;   // A: 128x32, 16 f32/thread
  const int brow = tid >> 3, bnc = (tid & 7) * 8;    // B: 32x64,   8 f32/thread

  for (int kt = 0; kt < 24; ++kt) {
    { // stage x tile 128x32 -> bf16
      const float* xp = x + (size_t)(m0 + arow) * EE + kt * 32 + akc;
      bf16* dst = As + arow * 40 + akc;
#pragma unroll
      for (int q = 0; q < 4; ++q) {
        v4f f = *(const v4f*)(xp + q * 4);
        dst[q*4+0]=(bf16)f.x; dst[q*4+1]=(bf16)f.y;
        dst[q*4+2]=(bf16)f.z; dst[q*4+3]=(bf16)f.w;
      }
    }
    { // stage W tile 32x64 -> bf16 transposed [n][k]
      const float* wp = w + (size_t)(kt * 32 + brow) * (3 * EE) + n0 + bnc;
      v4f f0 = *(const v4f*)wp;
      v4f f1 = *(const v4f*)(wp + 4);
      Bt[(bnc+0)*40 + brow]=(bf16)f0.x; Bt[(bnc+1)*40 + brow]=(bf16)f0.y;
      Bt[(bnc+2)*40 + brow]=(bf16)f0.z; Bt[(bnc+3)*40 + brow]=(bf16)f0.w;
      Bt[(bnc+4)*40 + brow]=(bf16)f1.x; Bt[(bnc+5)*40 + brow]=(bf16)f1.y;
      Bt[(bnc+6)*40 + brow]=(bf16)f1.z; Bt[(bnc+7)*40 + brow]=(bf16)f1.w;
    }
    __syncthreads();
    v16bf fa0 = frag_a(As + (mi0 + 0) * 16 * 40, 40);
    v16bf fa1 = frag_a(As + (mi0 + 1) * 16 * 40, 40);
    v16bf fb0 = frag_b(Bt + (nj0 + 0) * 16 * 40, 40);
    v16bf fb1 = frag_b(Bt + (nj0 + 1) * 16 * 40, 40);
    a00 = wmma_bf16(fa0, fb0, a00);
    a01 = wmma_bf16(fa0, fb1, a01);
    a10 = wmma_bf16(fa1, fb0, a10);
    a11 = wmma_bf16(fa1, fb1, a11);
    __syncthreads();
  }

  // transpose through LDS: Cs[n][m]
  const int cn = lane & 15, mhi = (lane >> 4) << 3;
#pragma unroll
  for (int r = 0; r < 8; ++r) {
    const int mA = (mi0 + 0) * 16 + r + mhi;
    const int mB = (mi0 + 1) * 16 + r + mhi;
    Cs[((nj0 + 0) * 16 + cn) * 136 + mA] = (bf16)a00[r];
    Cs[((nj0 + 1) * 16 + cn) * 136 + mA] = (bf16)a01[r];
    Cs[((nj0 + 0) * 16 + cn) * 136 + mB] = (bf16)a10[r];
    Cs[((nj0 + 1) * 16 + cn) * 136 + mB] = (bf16)a11[r];
  }
  __syncthreads();
  { // scatter per channel, 64B contiguous along s (runs never cross batch:
    // 3136 % 32 == 0, run length 32, start multiple of 32)
    const int cl = tid >> 2, sp = (tid & 3) * 32;
    const int c = n0 + cl;
    const int which = c / EE;
    const int rem = c - which * EE;
    const int h = rem / DD, d = rem - h * DD;
    const int gm = m0 + sp;
    const int b = gm / SS, sg = gm - b * SS;
    bf16* dst = qkvt + ((((size_t)which * BB + b) * HH + h) * DD + d) * SS + sg;
    const bf16* src = Cs + cl * 136 + sp;
#pragma unroll
    for (int q = 0; q < 4; ++q)
      *(v4u*)(dst + q * 8) = *(const v4u*)(src + q * 8);
  }
}

// ---------------------------------------------------------------------------
// Kernel 2: per (b,h) XCA: norms, gram=q@k^T (WMMA), softmax, out=attn@v (WMMA)
// 288 threads = 9 waves.
// ---------------------------------------------------------------------------
__global__ __launch_bounds__(288) void k_attn(const bf16* __restrict__ qkvt,
                                              const float* __restrict__ temp,
                                              bf16* __restrict__ aout) {
  __shared__ float gram[48 * 48];
  __shared__ float norms[96];
  __shared__ bf16  attnb[48 * 64];        // K zero-padded 48->64
  __shared__ bf16  vT[9 * 16 * 64];       // per-wave v^T tile [s][k(pad 64)]
  __shared__ bf16  ldsO[9 * 16 * 48];     // per-wave output bounce [s][d']

  const int tid  = threadIdx.x;
  const int wv   = tid >> 5;
  const int lane = tid & 31;
  const int bh = blockIdx.x;
  const int b = bh / HH, h = bh % HH;
  const size_t base = (((size_t)b * HH) + h) * (size_t)DD * SS;
  const size_t plane = (size_t)BB * HH * DD * SS;
  const bf16* qb = qkvt + base;
  const bf16* kb = qkvt + plane + base;
  const bf16* vb = qkvt + 2 * plane + base;

  // zero vT once: K rows 48..63 stay zero across all tiles
  for (int i = tid; i < 9 * 16 * 64; i += 288) vT[i] = (bf16)0.0f;

  // phase 1: L2 norms of q rows (0..47) and k rows (48..95) over s
  for (int r = wv; r < 96; r += 9) {
    const bf16* row = (r < 48) ? (qb + (size_t)r * SS) : (kb + (size_t)(r - 48) * SS);
    float s = 0.f;
    for (int c8 = lane * 8; c8 < SS; c8 += 32 * 8) {
      Pack8 p; p.q = *(const v4u*)(row + c8);
#pragma unroll
      for (int j = 0; j < 8; ++j) { float f = (float)p.h[j]; s += f * f; }
    }
    for (int off = 16; off; off >>= 1) s += __shfl_xor(s, off, 32);
    if (lane == 0) norms[r] = sqrtf(s);
  }

  // phase 2: gram = q @ k^T, 3x3 tiles of 16x16, one per wave, K = 3136
  {
    const int mi = wv / 3, nj = wv % 3;
    v8f acc = {};
    const bf16* ap = qb + (size_t)(mi * 16) * SS;
    const bf16* bp = kb + (size_t)(nj * 16) * SS;
    for (int ks = 0; ks < SS; ks += 32) {
      v16bf a  = frag_a(ap + ks, SS);   // rows contiguous in s
      v16bf bm = frag_b(bp + ks, SS);   // k stored [d][s] == [N][K]
      acc = wmma_bf16(a, bm, acc);
    }
    const int cn = lane & 15, mhi = (lane >> 4) << 3;
#pragma unroll
    for (int r = 0; r < 8; ++r)
      gram[(mi * 16 + r + mhi) * 48 + nj * 16 + cn] = acc[r];
  }
  __syncthreads();

  // phase 3: fold norms + temperature, softmax rows -> attnb (bf16, padded)
  const float tmph = temp[h];
  if (tid < 48) {
    const int i = tid;
    const float inv_nq = tmph / fmaxf(norms[i], XEPS);
    float mx = -3.4e38f;
    for (int j = 0; j < 48; ++j) {
      float v = gram[i * 48 + j] * inv_nq / fmaxf(norms[48 + j], XEPS);
      gram[i * 48 + j] = v;
      mx = fmaxf(mx, v);
    }
    float sum = 0.f;
    for (int j = 0; j < 48; ++j) {
      float e = __expf(gram[i * 48 + j] - mx);
      gram[i * 48 + j] = e; sum += e;
    }
    const float inv = 1.0f / sum;
    for (int j = 0; j < 48; ++j) attnb[i * 64 + j] = (bf16)(gram[i * 48 + j] * inv);
    for (int j = 48; j < 64; ++j) attnb[i * 64 + j] = (bf16)0.0f;
  }
  __syncthreads();

  // phase 4: out = attn @ v, s-tiles of 16 per wave; result -> [b,s,h,d'] bf16
  bf16* myvT = vT + wv * (16 * 64);
  bf16* myO  = ldsO + wv * (16 * 48);
  for (int nt = wv; nt < SS / 16; nt += 9) {
    // stage v[48 x 16] transposed into myvT[s][k]
    for (int d = lane; d < 48; d += 32) {
      const bf16* vr = vb + (size_t)d * SS + nt * 16;
      Pack8 p0, p1;
      p0.q = *(const v4u*)vr;
      p1.q = *(const v4u*)(vr + 8);
#pragma unroll
      for (int j = 0; j < 8; ++j) {
        myvT[j * 64 + d]       = p0.h[j];
        myvT[(j + 8) * 64 + d] = p1.h[j];
      }
    }
    asm volatile("" ::: "memory");   // DS ops are in-order per wave
    v8f o0 = {}, o1 = {}, o2 = {};
#pragma unroll
    for (int kk = 0; kk < 2; ++kk) {
      v16bf bm = frag_b(myvT + kk * 32, 64);
      v16bf a0 = frag_a(attnb + 0 * 16 * 64 + kk * 32, 64);
      v16bf a1 = frag_a(attnb + 1 * 16 * 64 + kk * 32, 64);
      v16bf a2 = frag_a(attnb + 2 * 16 * 64 + kk * 32, 64);
      o0 = wmma_bf16(a0, bm, o0);
      o1 = wmma_bf16(a1, bm, o1);
      o2 = wmma_bf16(a2, bm, o2);
    }
    const int cn = lane & 15, mhi = (lane >> 4) << 3;
#pragma unroll
    for (int r = 0; r < 8; ++r) {
      myO[cn * 48 + 0  + r + mhi] = (bf16)o0[r];
      myO[cn * 48 + 16 + r + mhi] = (bf16)o1[r];
      myO[cn * 48 + 32 + r + mhi] = (bf16)o2[r];
    }
    asm volatile("" ::: "memory");
    if (lane < 16) { // 96B contiguous per s row
      bf16* dst = aout + ((size_t)b * SS + nt * 16 + lane) * EE + h * DD;
      const bf16* src = myO + lane * 48;
#pragma unroll
      for (int j = 0; j < 6; ++j)
        *(v4u*)(dst + j * 8) = *(const v4u*)(src + j * 8);
    }
    asm volatile("" ::: "memory");
  }
}

// ---------------------------------------------------------------------------
// Kernel 3: y = aout(bf16) @ W_proj + b_proj -> f32.
// BM=128 BN=64, 2x2 tiles per wave. A tile is a pure bf16 global->LDS copy:
// use CDNA5 GLOBAL_LOAD_ASYNC_TO_LDS_B128 when the builtin is available.
// ---------------------------------------------------------------------------
__global__ __launch_bounds__(256) void k_proj(const bf16* __restrict__ a,
                                              const float* __restrict__ w,
                                              const float* __restrict__ bias,
                                              float* __restrict__ out) {
  __shared__ bf16 As[128 * 40];
  __shared__ bf16 Bt[64 * 40];
  const int tid = threadIdx.x;
  const int wv = tid >> 5, lane = tid & 31;
  const int m0 = blockIdx.x * 128, n0 = blockIdx.y * 64;
  const int mi0 = (wv & 3) * 2;
  const int nj0 = (wv >> 2) * 2;
  v8f a00 = {}, a01 = {}, a10 = {}, a11 = {};
  const int arow = tid >> 1, akc = (tid & 1) * 16;   // A: 128x32 bf16
  const int brow = tid >> 3, bnc = (tid & 7) * 8;    // B: 32x64 f32

  for (int kt = 0; kt < 24; ++kt) {
    { // stage A tile (already bf16): async LDS-DMA (b128) or plain 16B copies
      const bf16* src = a + (size_t)(m0 + arow) * EE + kt * 32 + akc;
      bf16* dst = As + arow * 40 + akc;
#if USE_ASYNC_LDS
      __builtin_amdgcn_global_load_async_to_lds_b128(
          (as1_v4i)(void*)src, (as3_v4i)(void*)dst, 0, 0);
      __builtin_amdgcn_global_load_async_to_lds_b128(
          (as1_v4i)(void*)(src + 8), (as3_v4i)(void*)(dst + 8), 0, 0);
#else
      *(v4u*)dst       = *(const v4u*)src;
      *(v4u*)(dst + 8) = *(const v4u*)(src + 8);
#endif
    }
    { // stage W tile 32x64 f32 -> bf16 transposed [n][k]
      const float* wp = w + (size_t)(kt * 32 + brow) * EE + n0 + bnc;
      v4f f0 = *(const v4f*)wp;
      v4f f1 = *(const v4f*)(wp + 4);
      Bt[(bnc+0)*40 + brow]=(bf16)f0.x; Bt[(bnc+1)*40 + brow]=(bf16)f0.y;
      Bt[(bnc+2)*40 + brow]=(bf16)f0.z; Bt[(bnc+3)*40 + brow]=(bf16)f0.w;
      Bt[(bnc+4)*40 + brow]=(bf16)f1.x; Bt[(bnc+5)*40 + brow]=(bf16)f1.y;
      Bt[(bnc+6)*40 + brow]=(bf16)f1.z; Bt[(bnc+7)*40 + brow]=(bf16)f1.w;
    }
#if USE_ASYNC_LDS
    __builtin_amdgcn_s_wait_asynccnt(0);
#endif
    __syncthreads();
    v16bf fa0 = frag_a(As + (mi0 + 0) * 16 * 40, 40);
    v16bf fa1 = frag_a(As + (mi0 + 1) * 16 * 40, 40);
    v16bf fb0 = frag_b(Bt + (nj0 + 0) * 16 * 40, 40);
    v16bf fb1 = frag_b(Bt + (nj0 + 1) * 16 * 40, 40);
    a00 = wmma_bf16(fa0, fb0, a00);
    a01 = wmma_bf16(fa0, fb1, a01);
    a10 = wmma_bf16(fa1, fb0, a10);
    a11 = wmma_bf16(fa1, fb1, a11);
    __syncthreads();
  }
  const int cn = lane & 15, mhi = (lane >> 4) << 3;
  const float bi0 = bias[n0 + (nj0 + 0) * 16 + cn];
  const float bi1 = bias[n0 + (nj0 + 1) * 16 + cn];
#pragma unroll
  for (int r = 0; r < 8; ++r) {
    const int mA = m0 + (mi0 + 0) * 16 + r + mhi;
    const int mB = m0 + (mi0 + 1) * 16 + r + mhi;
    out[(size_t)mA * EE + n0 + (nj0 + 0) * 16 + cn] = a00[r] + bi0;
    out[(size_t)mA * EE + n0 + (nj0 + 1) * 16 + cn] = a01[r] + bi1;
    out[(size_t)mB * EE + n0 + (nj0 + 0) * 16 + cn] = a10[r] + bi0;
    out[(size_t)mB * EE + n0 + (nj0 + 1) * 16 + cn] = a11[r] + bi1;
  }
}

extern "C" void kernel_launch(void* const* d_in, const int* in_sizes, int n_in,
                              void* d_out, int out_size, void* d_ws, size_t ws_size,
                              hipStream_t stream) {
  (void)in_sizes; (void)n_in; (void)out_size; (void)ws_size;
  const float* x      = (const float*)d_in[0];
  const float* w_qkv  = (const float*)d_in[1];
  const float* w_proj = (const float*)d_in[2];
  const float* b_proj = (const float*)d_in[3];
  const float* temp   = (const float*)d_in[4];

  bf16* qkvt = (bf16*)d_ws;                                  // 3*B*H*D*S bf16 (231MB)
  bf16* aout = qkvt + (size_t)3 * BB * HH * DD * SS;         // B*S*E bf16 (77MB)

  dim3 g1(392, 36);   // (B*S/128, 3E/64)
  dim3 g3(392, 12);   // (B*S/128,  E/64)
  k_qkv <<<g1, 256, 0, stream>>>(x, w_qkv, qkvt);
  k_attn<<<BB * HH, 288, 0, stream>>>(qkvt, temp, aout);
  k_proj<<<g3, 256, 0, stream>>>(aout, w_proj, b_proj, (float*)d_out);
}
